// RelAttLayer_23880018166447
// MI455X (gfx1250) — matrix operand order repaired and checked by reading, not compiled, not run on hardware
//
#include <hip/hip_runtime.h>
#include <hip/hip_bf16.h>
#include <stdint.h>

// ---------------------------------------------------------------------------
// RelAttLayer for MI455X (gfx1250, wave32, WMMA).
//
// Reference math:
//   a_e  = h[src]·u_s + he[e]·u_e + h[dst]·u_d          (u_* = Wa_* @ W_shared)
//   Hall = einsum('ni,rio->nro', h, W_rel)              (bf16 WMMA, f32 accum)
//   msg  = Hall[src, rel] * a_e ; out = relu(segment_sum(msg, dst))
// ---------------------------------------------------------------------------

typedef __attribute__((ext_vector_type(16))) __bf16 v16bf;
typedef __attribute__((ext_vector_type(8)))  float  v8f;

#define IN_DIM   64
#define OUT_DIM  64
#define NUM_RELS 16

union BfVec {                 // 16 bf16 = 32 bytes = two uint4 chunks
    v16bf v;
    uint4 q[2];
};

// ---- (a) u_s/u_e/u_d = W_attn chunks projected through W_shared -----------
__global__ void k_prep_u(const float* __restrict__ Wsh,   // [64,64] (out,in)
                         const float* __restrict__ Wa,    // [1,192]
                         float* __restrict__ u)           // [192]
{
    int i = threadIdx.x;                                  // 0..63 (input dim)
    float s0 = 0.f, s1 = 0.f, s2 = 0.f;
#pragma unroll 8
    for (int o = 0; o < OUT_DIM; ++o) {
        float w = Wsh[o * IN_DIM + i];
        s0 += Wa[o]       * w;
        s1 += Wa[64 + o]  * w;
        s2 += Wa[128 + o] * w;
    }
    u[i] = s0; u[64 + i] = s1; u[128 + i] = s2;
}

// ---- (b) fp32 -> bf16 conversion ------------------------------------------
__global__ void k_cvt_bf16(const float* __restrict__ src, __bf16* __restrict__ dst, int n)
{
    int i = blockIdx.x * blockDim.x + threadIdx.x;
    int stride = gridDim.x * blockDim.x;
    for (; i < n; i += stride) dst[i] = (__bf16)src[i];
}

// ---- (c) zero the accumulator (d_out) -------------------------------------
__global__ void k_zero(float* __restrict__ p, int n)
{
    int i = blockIdx.x * blockDim.x + threadIdx.x;
    int stride = gridDim.x * blockDim.x;
    for (; i < n; i += stride) p[i] = 0.f;
}

// ---- (d) Hall[n,r,o] = sum_i h[n,i] * W_rel[r,i,o]  (bf16 WMMA, f32 acc) --
// grid.x = relation, grid.y = group of 8 node-tiles; block = 256 (8 waves).
// Each wave computes one 16x64 output tile: K=64 -> 2 WMMA K-steps of 32,
// N=64 -> 4 tiles of 16.  B panel (64x64 bf16) staged in LDS, rows padded
// to 72 elements to break power-of-2 bank aliasing.
#define LDS_STRIDE 72
__global__ void k_gemm_rel(const __bf16* __restrict__ hbf,   // [N,64]
                           const __bf16* __restrict__ Wbf,   // [16,64,64]
                           float* __restrict__ Hall,         // [N,16,64]
                           int ntiles)
{
    __shared__ __bf16 Wlds[IN_DIM * LDS_STRIDE];

    const int r    = blockIdx.x;
    const int tid  = threadIdx.x;
    const int wave = tid >> 5;
    const int lane = tid & 31;

    // cooperative load of this relation's 64x64 B panel
    const __bf16* Wg = Wbf + (size_t)r * IN_DIM * OUT_DIM;
    for (int idx = tid; idx < IN_DIM * OUT_DIM; idx += 256) {
        int row = idx >> 6, col = idx & 63;
        Wlds[row * LDS_STRIDE + col] = Wg[idx];
    }
    __syncthreads();

    const int tile = blockIdx.y * 8 + wave;                // wave-uniform
    if (tile >= ntiles) return;

    const int m      = lane & 15;                          // A row (M)
    const int khalf8 = (lane >> 4) * 8;                    // A K sub-offset

    v8f acc[4];
#pragma unroll
    for (int t = 0; t < 4; ++t) acc[t] = (v8f)0.f;

#pragma unroll
    for (int k0 = 0; k0 < IN_DIM; k0 += 32) {
        // A 16x32 bf16: lane<16 -> K = k0+{0..7, 16..23}; lane>=16 -> +8
        BfVec a;
        const uint4* pa = (const uint4*)(hbf + ((size_t)tile * 16 + m) * IN_DIM
                                             + k0 + khalf8);
        a.q[0] = pa[0];        // K = k0+khalf8 .. +7
        a.q[1] = pa[2];        // K = k0+16+khalf8 .. +7   (+16 elems = 2 uint4)

#pragma unroll
        for (int t = 0; t < 4; ++t) {
            // B 32x16 bf16: lane holds column N = t*16+(lane&15); K in same order as A
            BfVec b;
#pragma unroll
            for (int s = 0; s < 16; ++s)
                b.v[s] = Wlds[(k0 + khalf8 + (s & 7) + ((s >> 3) << 4)) * LDS_STRIDE
                              + t * 16 + m];
            acc[t] = __builtin_amdgcn_wmma_f32_16x16x32_bf16(
                false, a.v, false, b.v, (short)0, acc[t], false, false);
        }
    }

    // C/D layout: VGPR j -> M=j (lanes 0-15) / M=j+8 (lanes 16-31); N = lane&15
    const int rowadd = (lane >> 4) * 8;
    const int col    = lane & 15;
#pragma unroll
    for (int t = 0; t < 4; ++t) {
#pragma unroll
        for (int j = 0; j < 8; ++j) {
            int row = j + rowadd;
            Hall[(((size_t)tile * 16 + row) * NUM_RELS + r) * OUT_DIM
                 + t * 16 + col] = acc[t][j];
        }
    }
}

// ---- (e) per-edge attention + message + scatter-add -----------------------
// One wave per edge; lane l owns elements {l, l+32} of each 64-vector.
__global__ void k_edge(const float* __restrict__ h,
                       const float* __restrict__ he,
                       const int*   __restrict__ src,
                       const int*   __restrict__ dst,
                       const int*   __restrict__ rel,
                       const float* __restrict__ u,      // [192]
                       const float* __restrict__ Hall,   // [N,16,64]
                       float* __restrict__ out,          // [N,64]
                       int E)
{
    const int lane = threadIdx.x & 31;
    const int wid  = (blockIdx.x * blockDim.x + threadIdx.x) >> 5;
    const int nw   = (gridDim.x * blockDim.x) >> 5;

    const float us0 = u[lane],       us1 = u[lane + 32];
    const float ue0 = u[64 + lane],  ue1 = u[96 + lane];
    const float ud0 = u[128 + lane], ud1 = u[160 + lane];

    for (int e = wid; e < E; e += nw) {
        const int s = src[e], d = dst[e], r = rel[e];
        const float* hs = h  + (size_t)s * 64;
        const float* hd = h  + (size_t)d * 64;
        const float* hq = he + (size_t)e * 64;

        float p = hs[lane] * us0 + hs[lane + 32] * us1
                + hq[lane] * ue0 + hq[lane + 32] * ue1
                + hd[lane] * ud0 + hd[lane + 32] * ud1;
#pragma unroll
        for (int off = 16; off; off >>= 1) p += __shfl_xor(p, off, 32);

        const float* msg = Hall + ((size_t)s * NUM_RELS + r) * 64;
        float* o = out + (size_t)d * 64;
        __hip_atomic_fetch_add(o + lane,      msg[lane]      * p,
                               __ATOMIC_RELAXED, __HIP_MEMORY_SCOPE_AGENT);
        __hip_atomic_fetch_add(o + lane + 32, msg[lane + 32] * p,
                               __ATOMIC_RELAXED, __HIP_MEMORY_SCOPE_AGENT);
    }
}

// ---- (f) in-place ReLU -----------------------------------------------------
__global__ void k_relu(float* __restrict__ p, int n)
{
    int i = blockIdx.x * blockDim.x + threadIdx.x;
    int stride = gridDim.x * blockDim.x;
    for (; i < n; i += stride) p[i] = fmaxf(p[i], 0.f);
}

// ---------------------------------------------------------------------------
extern "C" void kernel_launch(void* const* d_in, const int* in_sizes, int n_in,
                              void* d_out, int out_size, void* d_ws, size_t ws_size,
                              hipStream_t stream)
{
    const float* h   = (const float*)d_in[0];   // [N,64]
    const float* he  = (const float*)d_in[1];   // [E,64]
    const int*   src = (const int*)  d_in[2];   // [E]
    const int*   dst = (const int*)  d_in[3];   // [E]
    const int*   rel = (const int*)  d_in[4];   // [E]
    const float* Wsh = (const float*)d_in[5];   // [64,64]
    const float* Wa  = (const float*)d_in[6];   // [1,192]
    const float* Wr  = (const float*)d_in[7];   // [16,64,64]
    float* out = (float*)d_out;

    const int N = in_sizes[0] / IN_DIM;         // 50000
    const int E = in_sizes[2];                  // 500000
    const int nh = N * IN_DIM;                  // 3.2M
    const int nw = NUM_RELS * IN_DIM * OUT_DIM; // 65536
    const int ntiles = (N + 15) / 16;           // 3125 (exact)

    // workspace layout (all offsets 256B-aligned by construction)
    char* ws = (char*)d_ws;
    float*  u    = (float*)ws;                                     //    768 B
    __bf16* Wbf  = (__bf16*)(ws + 1024);                           // 131072 B
    __bf16* hbf  = (__bf16*)(ws + 1024 + 131072);                  //  6.4 MB
    float*  Hall = (float*)(ws + 1024 + 131072 + (size_t)nh * 2);  // 204.8 MB

    k_prep_u<<<1, 64, 0, stream>>>(Wsh, Wa, u);
    k_cvt_bf16<<<64, 256, 0, stream>>>(Wr, Wbf, nw);
    k_cvt_bf16<<<2048, 256, 0, stream>>>(h, hbf, nh);
    k_zero<<<2048, 256, 0, stream>>>(out, nh);

    dim3 ggrid(NUM_RELS, (ntiles + 7) / 8);
    k_gemm_rel<<<ggrid, 256, 0, stream>>>(hbf, Wbf, Hall, ntiles);

    k_edge<<<1024, 256, 0, stream>>>(h, he, src, dst, rel, u, Hall, out, E);
    k_relu<<<2048, 256, 0, stream>>>(out, nh);
}
